// GNNFeatureExtractor_68367289417949
// MI455X (gfx1250) — matrix operand order repaired and checked by reading, not compiled, not run on hardware
//
#include <hip/hip_runtime.h>

// ---------------------------------------------------------------------------
// GNN message-passing feature extractor for MI455X (gfx1250, wave32, WMMA).
// All GEMMs run as v_wmma_f32_16x16x32_bf16; one wave = one 16-row tile.
// ---------------------------------------------------------------------------

typedef __attribute__((ext_vector_type(16))) unsigned short v16us;
typedef __attribute__((ext_vector_type(16))) __bf16         v16bf;
typedef __attribute__((ext_vector_type(8)))  float          v8f;
typedef __attribute__((ext_vector_type(4)))  float          v4f;

union Frag {
    v16us us;
    v16bf bf;
    unsigned u[8];
};

__device__ __forceinline__ unsigned short f2bf(float f) {
    unsigned u = __float_as_uint(f);
    unsigned r = (u >> 16) & 1u;
    u += 0x7fffu + r;                 // round-to-nearest-even
    return (unsigned short)(u >> 16);
}
__device__ __forceinline__ float bf2f(unsigned short h) {
    return __uint_as_float(((unsigned)h) << 16);
}

__device__ __forceinline__ v8f wmma_bf16(const Frag& a, const Frag& b, v8f c) {
    return __builtin_amdgcn_wmma_f32_16x16x32_bf16(
        /*neg_a=*/false, a.bf, /*neg_b=*/false, b.bf,
        /*c_mod=*/(short)0, c, /*reuse_a=*/false, /*reuse_b=*/false);
}

// B operand: fragment-major bf16 weights, one contiguous 32B load per lane.
__device__ __forceinline__ Frag load_bfrag(const unsigned short* __restrict__ w,
                                           int kt, int nt, int ntiles, int lane) {
    Frag f;
    f.us = *(const v16us*)(w + (((long long)(kt * ntiles + nt)) << 9) + (lane << 4));
    return f;
}

// A operand built from an LDS tile (16 rows x 2^pitchShift bf16 cols).
// 16-bit A layout: lanes 0-15 -> rows; VGPR0-3 = K0..7, VGPR4-7 = K16..23;
// lanes 16-31 get +8. (CDNA5 ISA §7.12.2)
__device__ __forceinline__ Frag build_afrag(const unsigned short* lds,
                                            int pitchShift, int kbase, int lane) {
    Frag f;
    const unsigned short* row = lds + ((lane & 15) << pitchShift);
    int koff = kbase + ((lane >> 4) << 3);
#pragma unroll
    for (int e2 = 0; e2 < 8; ++e2) {
        int K = koff + ((e2 < 4) ? (e2 << 1) : (16 + ((e2 - 4) << 1)));
        f.u[e2] = *(const unsigned*)(row + K);
    }
    return f;
}

// Copy 64 f32 (one half-row) into a bf16 LDS row.
__device__ __forceinline__ void stage_row_half(unsigned short* dstRow,
                                               const float* __restrict__ srcRow,
                                               int c0) {
#pragma unroll
    for (int j = 0; j < 16; ++j) {
        v4f v = *(const v4f*)(srcRow + c0 + j * 4);
        unsigned lo = (unsigned)f2bf(v.x) | ((unsigned)f2bf(v.y) << 16);
        unsigned hi = (unsigned)f2bf(v.z) | ((unsigned)f2bf(v.w) << 16);
        unsigned long long p = (unsigned long long)lo | ((unsigned long long)hi << 32);
        *(unsigned long long*)(dstRow + c0 + j * 4) = p;
    }
}

// ---------------------------------------------------------------------------
// Weight prep: fp32 [K,N] row-major -> bf16 B-fragment layout
// frag[kt][nt][lane][e]; col=nt*16+(lane&15), h=lane>>4;
// K=kt*32+8h+e (e<8), K=kt*32+16+8h+(e-8) (e>=8). Zero-pad K beyond real K.
// ---------------------------------------------------------------------------
__global__ void prep_weight_kernel(const float* __restrict__ W, int K, int N,
                                   unsigned short* __restrict__ dst, int ntiles) {
    int f  = blockIdx.x;
    int kt = f / ntiles, nt = f % ntiles;
    int i0 = threadIdx.x * 2;
#pragma unroll
    for (int i = i0; i < i0 + 2; ++i) {
        int lane = i >> 4, e = i & 15;
        int kk = kt * 32 + ((lane >> 4) << 3) + ((e < 8) ? e : (8 + e));
        int nn = nt * 16 + (lane & 15);
        float v = (kk < K) ? W[(long long)kk * N + nn] : 0.0f;
        dst[(long long)f * 512 + i] = f2bf(v);
    }
}

__global__ void zero_f32(float* p, long long n) {
    long long i  = (long long)blockIdx.x * blockDim.x + threadIdx.x;
    long long st = (long long)gridDim.x * blockDim.x;
    for (; i < n; i += st) p[i] = 0.0f;
}

// ---------------------------------------------------------------------------
// Generic 2-layer MLP on flat rows (projection): din (<=32 padded) ->128->128
// One wave per 16-row tile.
// ---------------------------------------------------------------------------
__global__ __launch_bounds__(32) void proj_kernel(
    const float* __restrict__ in, int din,
    const unsigned short* __restrict__ W1f, const float* __restrict__ b1,
    const unsigned short* __restrict__ W2f, const float* __restrict__ b2,
    float* __restrict__ out) {
    __shared__ alignas(16) unsigned short blk[16 * 32];
    __shared__ alignas(16) unsigned short hid[16 * 128];
    int lane = threadIdx.x;
    long long row0 = (long long)blockIdx.x << 4;

    // zero-pad input tile, then fill real columns
    for (int i = lane; i < 256; i += 32) ((unsigned*)blk)[i] = 0u;
    for (int i = lane; i < 16 * din; i += 32) {
        int r = i / din, c = i % din;
        blk[r * 32 + c] = f2bf(in[(row0 + r) * din + c]);
    }

    int col0 = lane & 15, mrow = (lane >> 4) * 8;

    // GEMM1 (K=32 padded) -> hidden 128
    v8f acc[8] = {};
    {
        Frag a = build_afrag(blk, /*pitchShift=*/5, 0, lane);
#pragma unroll
        for (int nt = 0; nt < 8; ++nt) {
            Frag b = load_bfrag(W1f, 0, nt, 8, lane);
            acc[nt] = wmma_bf16(a, b, acc[nt]);
        }
    }
#pragma unroll
    for (int nt = 0; nt < 8; ++nt) {
        float bb = b1[nt * 16 + col0];
#pragma unroll
        for (int r = 0; r < 8; ++r) {
            float h = acc[nt][r] + bb;
            h = h > 0.0f ? h : 0.0f;
            hid[(mrow + r) * 128 + nt * 16 + col0] = f2bf(h);
        }
    }
    // GEMM2 128->128
    v8f acc2[8] = {};
    for (int kt = 0; kt < 4; ++kt) {
        Frag a = build_afrag(hid, 7, kt * 32, lane);
#pragma unroll
        for (int nt = 0; nt < 8; ++nt) {
            Frag b = load_bfrag(W2f, kt, nt, 8, lane);
            acc2[nt] = wmma_bf16(a, b, acc2[nt]);
        }
    }
#pragma unroll
    for (int nt = 0; nt < 8; ++nt) {
        float bb = b2[nt * 16 + col0];
#pragma unroll
        for (int r = 0; r < 8; ++r)
            out[(row0 + mrow + r) * 128 + nt * 16 + col0] = acc2[nt][r] + bb;
    }
}

// ---------------------------------------------------------------------------
// Fused edge layer: psi([e,x_i,x_j]) -> e_out (+residual store),
// phi([x_i,x_j,e_out]) -> messages, scatter-add into agg.
// One wave per 16-edge tile; e updated in place.
// ---------------------------------------------------------------------------
__global__ __launch_bounds__(32) void edge_layer_kernel(
    const float* __restrict__ xh, float* __restrict__ eh,
    const int* __restrict__ senders, const int* __restrict__ receivers,
    const unsigned short* __restrict__ psiW1, const float* __restrict__ psib1,
    const unsigned short* __restrict__ psiW2, const float* __restrict__ psib2,
    const unsigned short* __restrict__ phiW1, const float* __restrict__ phib1,
    const unsigned short* __restrict__ phiW2, const float* __restrict__ phib2,
    float* __restrict__ agg, int Nn, int Ee, int dout, int residual) {
    __shared__ alignas(16) unsigned short blk[3][16 * 128]; // e | x_i | x_j
    __shared__ alignas(16) unsigned short hid[16 * 128];
    int lane = threadIdx.x;
    int tpb  = Ee >> 4;
    int b    = blockIdx.x / tpb;
    int row0 = (blockIdx.x % tpb) << 4;

    const float* xb = xh + (long long)b * Nn * 128;
    float*       eb = eh + (long long)b * Ee * 128;
    float*       ab = agg + (long long)b * Nn * 128;

    // --- stage: e rows + gathered x rows (2 lanes per row, bf16 into LDS) ---
    {
        int row = lane >> 1, c0 = (lane & 1) * 64;
        stage_row_half(&blk[0][row * 128], eb + (long long)(row0 + row) * 128, c0);
        int ri = receivers[row0 + row];
        stage_row_half(&blk[1][row * 128], xb + (long long)ri * 128, c0);
        int si = senders[row0 + row];
        stage_row_half(&blk[2][row * 128], xb + (long long)si * 128, c0);
    }
    int col0 = lane & 15, mrow = (lane >> 4) * 8;
    int rIdx[8];
#pragma unroll
    for (int r = 0; r < 8; ++r) rIdx[r] = receivers[row0 + mrow + r];

    int nt2 = dout >> 4;  // output col tiles (4 or 8)
    int ktE = dout >> 5;  // e_out K tiles for phi (2 or 4)

    // ---------------- psi GEMM1: [e|x_i|x_j] (K=384) -> hidden ----------------
    v8f acc[8] = {};
    for (int kt = 0; kt < 12; ++kt) {
        Frag a = build_afrag(&blk[kt >> 2][0], 7, (kt & 3) * 32, lane);
#pragma unroll
        for (int nt = 0; nt < 8; ++nt) {
            Frag bf = load_bfrag(psiW1, kt, nt, 8, lane);
            acc[nt] = wmma_bf16(a, bf, acc[nt]);
        }
    }
#pragma unroll
    for (int nt = 0; nt < 8; ++nt) {
        float bb = psib1[nt * 16 + col0];
#pragma unroll
        for (int r = 0; r < 8; ++r) {
            float h = acc[nt][r] + bb;
            h = h > 0.0f ? h : 0.0f;
            hid[(mrow + r) * 128 + nt * 16 + col0] = f2bf(h);
        }
    }
    // ---------------- psi GEMM2: hidden -> e_out ----------------
    v8f acc2[8] = {};
    for (int kt = 0; kt < 4; ++kt) {
        Frag a = build_afrag(hid, 7, kt * 32, lane);
        for (int nt = 0; nt < nt2; ++nt) {
            Frag bf = load_bfrag(psiW2, kt, nt, nt2, lane);
            acc2[nt] = wmma_bf16(a, bf, acc2[nt]);
        }
    }
    // store e (+residual), stash raw e_out into blk[0] for phi
    for (int nt = 0; nt < nt2; ++nt) {
        float bb = psib2[nt * 16 + col0];
#pragma unroll
        for (int r = 0; r < 8; ++r) {
            int M = mrow + r, col = nt * 16 + col0;
            float eo = acc2[nt][r] + bb;
            float st = eo;
            if (residual) st += bf2f(blk[0][M * 128 + col]);
            eb[(long long)(row0 + M) * 128 + col] = st;
            blk[0][M * 128 + col] = f2bf(eo);
        }
    }

    // ---------------- phi GEMM1: [x_i|x_j|e_out] -> hidden ----------------
    {
        v8f z = {};
#pragma unroll
        for (int nt = 0; nt < 8; ++nt) acc[nt] = z;
    }
    int ktphi = 8 + ktE;
    for (int kt = 0; kt < ktphi; ++kt) {
        const unsigned short* src;
        int kb;
        if (kt < 4)      { src = &blk[1][0]; kb = kt * 32; }
        else if (kt < 8) { src = &blk[2][0]; kb = (kt - 4) * 32; }
        else             { src = &blk[0][0]; kb = (kt - 8) * 32; }
        Frag a = build_afrag(src, 7, kb, lane);
#pragma unroll
        for (int nt = 0; nt < 8; ++nt) {
            Frag bf = load_bfrag(phiW1, kt, nt, 8, lane);
            acc[nt] = wmma_bf16(a, bf, acc[nt]);
        }
    }
#pragma unroll
    for (int nt = 0; nt < 8; ++nt) {
        float bb = phib1[nt * 16 + col0];
#pragma unroll
        for (int r = 0; r < 8; ++r) {
            float h = acc[nt][r] + bb;
            h = h > 0.0f ? h : 0.0f;
            hid[(mrow + r) * 128 + nt * 16 + col0] = f2bf(h);
        }
    }
    // ---------------- phi GEMM2: hidden -> messages; scatter-add ----------------
    {
        v8f z = {};
#pragma unroll
        for (int nt = 0; nt < 8; ++nt) acc2[nt] = z;
    }
    for (int kt = 0; kt < 4; ++kt) {
        Frag a = build_afrag(hid, 7, kt * 32, lane);
        for (int nt = 0; nt < nt2; ++nt) {
            Frag bf = load_bfrag(phiW2, kt, nt, nt2, lane);
            acc2[nt] = wmma_bf16(a, bf, acc2[nt]);
        }
    }
    for (int nt = 0; nt < nt2; ++nt) {
        float bb = phib2[nt * 16 + col0];
#pragma unroll
        for (int r = 0; r < 8; ++r) {
            float m = acc2[nt][r] + bb;
            unsafeAtomicAdd(&ab[(long long)rIdx[r] * 128 + nt * 16 + col0], m);
        }
    }
}

// ---------------------------------------------------------------------------
// Node update: node_update([x, agg]) -> x_out (+ optional residual).
// ---------------------------------------------------------------------------
__global__ __launch_bounds__(32) void node_layer_kernel(
    const float* __restrict__ xh, const float* __restrict__ agg,
    float* __restrict__ xout,
    const unsigned short* __restrict__ W1f, const float* __restrict__ b1,
    const unsigned short* __restrict__ W2f, const float* __restrict__ b2,
    int Nn, int dagg, int dout, int residual) {
    __shared__ alignas(16) unsigned short blk[2][16 * 128]; // x | agg
    __shared__ alignas(16) unsigned short hid[16 * 128];
    int lane = threadIdx.x;
    int tpb  = Nn >> 4;
    int b    = blockIdx.x / tpb;
    int row0 = (blockIdx.x % tpb) << 4;

    const float* xb = xh  + (long long)b * Nn * 128;
    const float* ab = agg + (long long)b * Nn * 128;
    float*       ob = xout + (long long)b * Nn * 128;

    {
        int row = lane >> 1, c0 = (lane & 1) * 64;
        stage_row_half(&blk[0][row * 128], xb + (long long)(row0 + row) * 128, c0);
        stage_row_half(&blk[1][row * 128], ab + (long long)(row0 + row) * 128, c0);
    }
    int col0 = lane & 15, mrow = (lane >> 4) * 8;
    int ktA = 4 + (dagg >> 5);
    int nt2 = dout >> 4;

    v8f acc[8] = {};
    for (int kt = 0; kt < ktA; ++kt) {
        const unsigned short* src = (kt < 4) ? &blk[0][0] : &blk[1][0];
        int kb = (kt < 4) ? kt * 32 : (kt - 4) * 32;
        Frag a = build_afrag(src, 7, kb, lane);
#pragma unroll
        for (int nt = 0; nt < 8; ++nt) {
            Frag bf = load_bfrag(W1f, kt, nt, 8, lane);
            acc[nt] = wmma_bf16(a, bf, acc[nt]);
        }
    }
#pragma unroll
    for (int nt = 0; nt < 8; ++nt) {
        float bb = b1[nt * 16 + col0];
#pragma unroll
        for (int r = 0; r < 8; ++r) {
            float h = acc[nt][r] + bb;
            h = h > 0.0f ? h : 0.0f;
            hid[(mrow + r) * 128 + nt * 16 + col0] = f2bf(h);
        }
    }
    v8f acc2[8] = {};
    for (int kt = 0; kt < 4; ++kt) {
        Frag a = build_afrag(hid, 7, kt * 32, lane);
        for (int nt = 0; nt < nt2; ++nt) {
            Frag bf = load_bfrag(W2f, kt, nt, nt2, lane);
            acc2[nt] = wmma_bf16(a, bf, acc2[nt]);
        }
    }
    for (int nt = 0; nt < nt2; ++nt) {
        float bb = b2[nt * 16 + col0];
#pragma unroll
        for (int r = 0; r < 8; ++r) {
            int M = mrow + r, col = nt * 16 + col0;
            float v = acc2[nt][r] + bb;
            if (residual) v += bf2f(blk[0][M * 128 + col]);
            ob[(long long)(row0 + M) * 128 + col] = v;
        }
    }
}

// ---------------------------------------------------------------------------
// Two-stage mean over rows: partial sums -> f32 atomics into pre-zeroed out.
// 64 active columns; src pitch 128.
// ---------------------------------------------------------------------------
__global__ void mean_kernel(const float* __restrict__ src, long long rowsPerBatch,
                            float scale, float* __restrict__ dst, int dstOff) {
    int b = blockIdx.y;
    const float* s = src + (long long)b * rowsPerBatch * 128;
    int col = threadIdx.x & 63;
    int rg  = threadIdx.x >> 6;  // 0..3
    float sum = 0.0f;
    for (long long r = (long long)blockIdx.x * 4 + rg; r < rowsPerBatch;
         r += (long long)gridDim.x * 4)
        sum += s[r * 128 + col];
    __shared__ float red[256];
    red[threadIdx.x] = sum;
    __syncthreads();
    if (rg == 0) {
        float t = red[col] + red[col + 64] + red[col + 128] + red[col + 192];
        unsafeAtomicAdd(&dst[b * 128 + dstOff + col], t * scale);
    }
}

// ---------------------------------------------------------------------------
// Host side
// ---------------------------------------------------------------------------
namespace {
constexpr int Bb = 4, Nn = 10000, Ee = 80000;

// d_in layout: jax pytree flatten order (sorted dict keys).
enum {
    IN_X = 0, IN_E = 1, IN_EIDX = 2,
    EP_W1 = 3, EP_W2 = 4, EP_B1 = 5, EP_B2 = 6,
    FN_NU_W1 = 7, FN_NU_W2 = 8, FN_NU_B1 = 9, FN_NU_B2 = 10,
    FN_PHI_W1 = 11, FN_PHI_W2 = 12, FN_PHI_B1 = 13, FN_PHI_B2 = 14,
    FN_PSI_W1 = 15, FN_PSI_W2 = 16, FN_PSI_B1 = 17, FN_PSI_B2 = 18,
    L_BASE = 19,   // layer l: base = 19 + 12*l; nu W1,W2,b1,b2 / phi / psi
    NP_W1 = 43, NP_W2 = 44, NP_B1 = 45, NP_B2 = 46,
    N_PARAMS = 47
};

struct WS { int idx; int K; int N; };
// frag-buffer order (weights only):
// 0 np.W1  1 np.W2  2 ep.W1  3 ep.W2
// per layer l: 4+6l psiW1, 5+6l psiW2, 6+6l phiW1, 7+6l phiW2, 8+6l nuW1, 9+6l nuW2
// final: 16..21
const WS kSpecs[22] = {
    {NP_W1, 16, 128}, {NP_W2, 128, 128}, {EP_W1, 8, 128}, {EP_W2, 128, 128},
    {L_BASE + 8, 384, 128}, {L_BASE + 9, 128, 128},
    {L_BASE + 4, 384, 128}, {L_BASE + 5, 128, 128},
    {L_BASE + 0, 256, 128}, {L_BASE + 1, 128, 128},
    {L_BASE + 20, 384, 128}, {L_BASE + 21, 128, 128},
    {L_BASE + 16, 384, 128}, {L_BASE + 17, 128, 128},
    {L_BASE + 12, 256, 128}, {L_BASE + 13, 128, 128},
    {FN_PSI_W1, 384, 128}, {FN_PSI_W2, 128, 64},
    {FN_PHI_W1, 320, 128}, {FN_PHI_W2, 128, 64},
    {FN_NU_W1, 192, 128}, {FN_NU_W2, 128, 64},
};

inline size_t alignUp(size_t v, size_t a) { return (v + a - 1) & ~(a - 1); }
}  // namespace

extern "C" void kernel_launch(void* const* d_in, const int* in_sizes, int n_in,
                              void* d_out, int out_size, void* d_ws, size_t ws_size,
                              hipStream_t stream) {
    (void)in_sizes; (void)out_size;
    if (n_in < N_PARAMS) return;

    const float* x    = (const float*)d_in[IN_X];
    const float* e    = (const float*)d_in[IN_E];
    const int*   eidx = (const int*)d_in[IN_EIDX];
    const int* senders   = eidx;
    const int* receivers = eidx + Ee;
    float* out = (float*)d_out;

    // ---- workspace layout ----
    unsigned short* wf = (unsigned short*)d_ws;
    size_t woff[22];
    size_t cur = 0;
    for (int i = 0; i < 22; ++i) {
        woff[i] = cur;
        int kt = (kSpecs[i].K + 31) / 32, nt = kSpecs[i].N / 16;
        cur += (size_t)kt * nt * 512;
    }
    size_t byteCur = alignUp(cur * sizeof(unsigned short), 256);
    char* wsB = (char*)d_ws;
    const size_t xBytes = (size_t)Bb * Nn * 128 * sizeof(float);
    const size_t eBytes = (size_t)Bb * Ee * 128 * sizeof(float);
    float* xA   = (float*)(wsB + byteCur); byteCur += xBytes;
    float* xBuf = (float*)(wsB + byteCur); byteCur += xBytes;
    float* agg  = (float*)(wsB + byteCur); byteCur += xBytes;
    float* eH   = (float*)(wsB + byteCur); byteCur += eBytes;
    if (byteCur > ws_size) return;  // workspace too small; nothing sane to do

    // ---- weight prep (bf16 fragment layout) ----
    for (int i = 0; i < 22; ++i) {
        int kt = (kSpecs[i].K + 31) / 32, nt = kSpecs[i].N / 16;
        prep_weight_kernel<<<kt * nt, 256, 0, stream>>>(
            (const float*)d_in[kSpecs[i].idx], kSpecs[i].K, kSpecs[i].N,
            wf + woff[i], nt);
    }

    // ---- input projections ----
    proj_kernel<<<(Bb * Nn) / 16, 32, 0, stream>>>(
        x, 16, wf + woff[0], (const float*)d_in[NP_B1],
        wf + woff[1], (const float*)d_in[NP_B2], xA);
    proj_kernel<<<(Bb * Ee) / 16, 32, 0, stream>>>(
        e, 8, wf + woff[2], (const float*)d_in[EP_B1],
        wf + woff[3], (const float*)d_in[EP_B2], eH);

    float* xcur = xA;
    float* xnxt = xBuf;
    const long long aggN = (long long)Bb * Nn * 128;

    // ---- residual message-passing layers ----
    for (int l = 0; l < 2; ++l) {
        int pb = L_BASE + 12 * l;
        zero_f32<<<1024, 256, 0, stream>>>(agg, aggN);
        edge_layer_kernel<<<(Bb * Ee) / 16, 32, 0, stream>>>(
            xcur, eH, senders, receivers,
            wf + woff[4 + 6 * l], (const float*)d_in[pb + 10],
            wf + woff[5 + 6 * l], (const float*)d_in[pb + 11],
            wf + woff[6 + 6 * l], (const float*)d_in[pb + 6],
            wf + woff[7 + 6 * l], (const float*)d_in[pb + 7],
            agg, Nn, Ee, /*dout=*/128, /*residual=*/1);
        node_layer_kernel<<<(Bb * Nn) / 16, 32, 0, stream>>>(
            xcur, agg, xnxt,
            wf + woff[8 + 6 * l], (const float*)d_in[pb + 2],
            wf + woff[9 + 6 * l], (const float*)d_in[pb + 3],
            Nn, /*dagg=*/128, /*dout=*/128, /*residual=*/1);
        float* t = xcur; xcur = xnxt; xnxt = t;
    }

    // ---- final layer (no residual, dout = 64) ----
    zero_f32<<<1024, 256, 0, stream>>>(agg, aggN);
    edge_layer_kernel<<<(Bb * Ee) / 16, 32, 0, stream>>>(
        xcur, eH, senders, receivers,
        wf + woff[16], (const float*)d_in[FN_PSI_B1],
        wf + woff[17], (const float*)d_in[FN_PSI_B2],
        wf + woff[18], (const float*)d_in[FN_PHI_B1],
        wf + woff[19], (const float*)d_in[FN_PHI_B2],
        agg, Nn, Ee, /*dout=*/64, /*residual=*/0);
    node_layer_kernel<<<(Bb * Nn) / 16, 32, 0, stream>>>(
        xcur, agg, xnxt,
        wf + woff[20], (const float*)d_in[FN_NU_B1],
        wf + woff[21], (const float*)d_in[FN_NU_B2],
        Nn, /*dagg=*/64, /*dout=*/64, /*residual=*/0);

    // ---- means -> output (B x 128 = [x_mean(64) | e_mean(64)]) ----
    zero_f32<<<1, 512, 0, stream>>>(out, (long long)Bb * 128);
    mean_kernel<<<dim3(64, Bb), 256, 0, stream>>>(xnxt, Nn, 1.0f / Nn, out, 0);
    mean_kernel<<<dim3(64, Bb), 256, 0, stream>>>(eH, Ee, 1.0f / Ee, out, 64);
}
